// TNODE_19679540150362
// MI455X (gfx1250) — matrix operand order repaired and checked by reading, not compiled, not run on hardware
//
#include <hip/hip_runtime.h>
#include <hip/hip_bf16.h>
#include <math.h>

typedef __attribute__((ext_vector_type(16))) _Float16 v16h;
typedef __attribute__((ext_vector_type(8)))  float    v8f;

#define N_CELLS 16384
#define N_INT   4096
#define N_LAT   5
#define N_ODEH  25
#define N_VH    128
#define EPSF    1e-8f

// ---- workspace layout (units: floats) ----
static constexpr long OFF_ACC   = 0;                               // 8 accumulators
static constexpr long OFF_THETA = 64;                              // 4096
static constexpr long OFF_LT    = OFF_THETA + N_INT;               // log(theta+eps)
static constexpr long OFF_LG    = OFF_LT + N_INT;                  // lgamma(theta)
static constexpr long OFF_H     = 16384;                           // h: 16384x128
static constexpr long OFF_T     = OFF_H + (long)N_CELLS * N_VH;    // T
static constexpr long OFF_TS    = OFF_T + N_CELLS;                 // sorted T
static constexpr long OFF_IDX   = OFF_TS + N_CELLS;                // argsort (int)
static constexpr long OFF_Z     = OFF_IDX + N_CELLS;               // z: 16384x5
static constexpr long OFF_PZ    = OFF_Z + (long)N_CELLS * N_LAT;   // pred_z
static constexpr long OFF_HDS   = OFF_PZ + (long)N_CELLS * N_LAT;  // swizzled HD A-frags (f16)
                                                                   // [1024 tiles][2 mt][4 ks][32 lane][16]
static constexpr long OFF_W2S   = OFF_HDS + (long)N_CELLS * N_VH;  // dec_W2 swizzled f16
static constexpr long OFF_W1S   = OFF_W2S + (long)N_VH * N_INT / 2;// enc_W1 swizzled f16

__device__ __forceinline__ float eluf(float v) { return v > 0.f ? v : expm1f(v); }

// ---------------- init: accumulators + per-gene theta terms -----------------
__global__ __launch_bounds__(256) void k_init(const float* __restrict__ disp,
                                              float* __restrict__ ws)
{
  int t = blockIdx.x * 256 + threadIdx.x;
  if (t < 8) ws[OFF_ACC + t] = 0.f;
  if (t < N_INT) {
    float th = __expf(disp[t]);
    ws[OFF_THETA + t] = th;
    ws[OFF_LT + t]    = logf(th + EPSF);
    ws[OFF_LG + t]    = lgammaf(th);
  }
}

// ---------------- pre-swizzle enc_W1 into B-fragment order -------------------
// frag = kc*8 + nt  (kc: 128 k-chunks of 32, nt: 8 col tiles of 16)
__global__ __launch_bounds__(256) void k_swz_w1(const float* __restrict__ w1,
                                                _Float16* __restrict__ w1s)
{
  int t = blockIdx.x * 256 + threadIdx.x;     // 1024 frags * 32 lanes
  int frag = t >> 5, lane = t & 31;
  int kc = frag >> 3, nt = frag & 7;
  int ml = lane & 15, g = lane >> 4;
  int n = nt * 16 + ml;
  v16h f;
#pragma unroll
  for (int v = 0; v < 8; ++v) {               // B 32x16 f16 layout
    int k = kc * 32 + ((v >> 2) << 4) + (g << 3) + ((v & 3) << 1);
    f[2 * v]     = (_Float16)w1[(size_t)k * N_VH + n];
    f[2 * v + 1] = (_Float16)w1[(size_t)(k + 1) * N_VH + n];
  }
  *(v16h*)(w1s + ((size_t)frag * 32 + lane) * 16) = f;
}

// ---------------- pre-swizzle dec_W2 into B-fragment order -------------------
// frag = nt*4 + kc  (nt: 256 col tiles of 16, kc: 4 k-chunks of 32)
__global__ __launch_bounds__(256) void k_swz_w2(const float* __restrict__ w2,
                                                _Float16* __restrict__ w2s)
{
  int t = blockIdx.x * 256 + threadIdx.x;     // 1024 frags * 32 lanes
  int frag = t >> 5, lane = t & 31;
  int nt = frag >> 2, kc = frag & 3;
  int ml = lane & 15, g = lane >> 4;
  int n = nt * 16 + ml;
  v16h f;
#pragma unroll
  for (int v = 0; v < 8; ++v) {
    int k = kc * 32 + ((v >> 2) << 4) + (g << 3) + ((v & 3) << 1);
    f[2 * v]     = (_Float16)w2[(size_t)k * N_INT + n];
    f[2 * v + 1] = (_Float16)w2[(size_t)(k + 1) * N_INT + n];
  }
  *(v16h*)(w2s + ((size_t)frag * 32 + lane) * 16) = f;
}

// ---------------- encoder: h = elu(x @ enc_W1 + b1), WMMA f16 ----------------
// one wave = 16 rows x 64 cols strip (A fragment reused over 4 n-tiles)
__global__ __launch_bounds__(128) void k_encoder(const float* __restrict__ x,
                                                 const _Float16* __restrict__ w1s,
                                                 const float* __restrict__ b1,
                                                 float* __restrict__ h)
{
  const int lane   = threadIdx.x & 31;
  const int wave   = threadIdx.x >> 5;
  const int waveId = blockIdx.x * 4 + wave;   // 2048 waves total
  const int mt = waveId >> 1;                 // 1024 row tiles
  const int nh = waveId & 1;                  // which 64-col half
  const int ml = lane & 15;
  const int g  = lane >> 4;
  const int m  = (mt << 4) + ml;

  v8f zero = {0.f, 0.f, 0.f, 0.f, 0.f, 0.f, 0.f, 0.f};
  v8f c[4];
#pragma unroll
  for (int j = 0; j < 4; ++j) c[j] = zero;

  const float* xrow = x + (size_t)m * N_INT;
  for (int kb = 0; kb < N_INT; kb += 32) {
    const int kc = kb >> 5;
    v16h a;
#pragma unroll
    for (int v = 0; v < 8; ++v) {     // A 16x32 f16 layout (ISA 7.12.2)
      int ka = kb + ((v >> 2) << 4) + (g << 3) + ((v & 3) << 1);
      float2 xa = *(const float2*)(xrow + ka);
      a[2 * v]     = (_Float16)xa.x;
      a[2 * v + 1] = (_Float16)xa.y;
    }
#pragma unroll
    for (int j = 0; j < 4; ++j) {
      const int nt = (nh << 2) + j;
      v16h b = *(const v16h*)(w1s + (((size_t)kc * 8 + nt) * 32 + lane) * 16);
      c[j] = __builtin_amdgcn_wmma_f32_16x16x32_f16(false, a, false, b,
                                                    (short)0, c[j], false, false);
    }
  }
#pragma unroll
  for (int j = 0; j < 4; ++j) {
    const int n = (nh << 6) + (j << 4) + ml;
    const float bn = b1[n];
#pragma unroll
    for (int v = 0; v < 8; ++v) {     // C layout: row = v + 8*g, col = ml
      int row = (mt << 4) + v + (g << 3);
      h[(size_t)row * N_VH + n] = eluf(c[j][v] + bn);
    }
  }
}

// ---------------- per-cell stats: qz mean/logvar, T, z, KL -------------------
__global__ __launch_bounds__(256) void k_stats(const float* __restrict__ h,
                                               const float* __restrict__ w2,
                                               const float* __restrict__ b2,
                                               const float* __restrict__ w3,
                                               const float* __restrict__ b3,
                                               const float* __restrict__ eps,
                                               float* __restrict__ Tarr,
                                               float* __restrict__ zarr,
                                               float* __restrict__ acc)
{
  int i = blockIdx.x * blockDim.x + threadIdx.x;
  if (i >= N_CELLS) return;
  float s[10];
#pragma unroll
  for (int o = 0; o < 10; ++o) s[o] = 0.f;
  float sT = 0.f;
  const float* hr = h + (size_t)i * N_VH;
  for (int cc = 0; cc < N_VH; ++cc) {
    float hv = hr[cc];
#pragma unroll
    for (int o = 0; o < 10; ++o) s[o] += hv * w2[cc * 10 + o];
    sT += hv * w3[cc];
  }
  Tarr[i] = 1.f / (1.f + __expf(-(sT + b3[0])));
  float kl = 0.f;
#pragma unroll
  for (int k = 0; k < 5; ++k) {
    float mean = s[k] + b2[k];
    float lv   = s[5 + k] + b2[5 + k];
    zarr[(size_t)i * 5 + k] = eps[(size_t)i * 5 + k] * __expf(0.5f * lv) + mean;
    kl += -0.5f * lv + (__expf(lv) + mean * mean) * 0.5f - 0.5f;
  }
  atomicAdd(acc + 0, kl);
}

// ---------------- stable argsort by ranking (O(N^2), parallel) ---------------
__global__ __launch_bounds__(256) void k_rank(const float* __restrict__ Tarr,
                                              int* __restrict__ idx,
                                              float* __restrict__ Ts)
{
  __shared__ int red[256];
  const int i = blockIdx.x;
  const float Ti = Tarr[i];
  int cnt = 0;
  for (int j = threadIdx.x; j < N_CELLS; j += 256) {
    float Tj = Tarr[j];
    cnt += (Tj < Ti) || (Tj == Ti && j < i);
  }
  red[threadIdx.x] = cnt;
  __syncthreads();
  for (int off = 128; off; off >>= 1) {
    if (threadIdx.x < off) red[threadIdx.x] += red[threadIdx.x + off];
    __syncthreads();
  }
  if (threadIdx.x == 0) { int r = red[0]; idx[r] = i; Ts[r] = Ti; }
}

// ---------------- sequential ODE scan: single wave ---------------------------
__global__ __launch_bounds__(32) void k_ode(const float* __restrict__ w1,
                                            const float* __restrict__ bb1,
                                            const float* __restrict__ w2,
                                            const float* __restrict__ bb2,
                                            const float* __restrict__ Ts,
                                            const int* __restrict__ idx,
                                            const float* __restrict__ zarr,
                                            float* __restrict__ pz)
{
  __shared__ float sT[2048];
  const int lane = threadIdx.x;
  float w10 = 0, w11 = 0, w12 = 0, w13 = 0, w14 = 0, b1j = 0;
  float w20 = 0, w21 = 0, w22 = 0, w23 = 0, w24 = 0;
  if (lane < N_ODEH) {
    w10 = w1[0 * N_ODEH + lane]; w11 = w1[1 * N_ODEH + lane];
    w12 = w1[2 * N_ODEH + lane]; w13 = w1[3 * N_ODEH + lane];
    w14 = w1[4 * N_ODEH + lane];
    b1j = bb1[lane];
    w20 = w2[lane * 5 + 0]; w21 = w2[lane * 5 + 1]; w22 = w2[lane * 5 + 2];
    w23 = w2[lane * 5 + 3]; w24 = w2[lane * 5 + 4];
  }
  const float b20 = bb2[0], b21 = bb2[1], b22 = bb2[2], b23 = bb2[3], b24 = bb2[4];
  const int i0 = idx[0];
  float z0 = zarr[(size_t)i0 * 5 + 0], z1 = zarr[(size_t)i0 * 5 + 1],
        z2 = zarr[(size_t)i0 * 5 + 2], z3 = zarr[(size_t)i0 * 5 + 3],
        z4 = zarr[(size_t)i0 * 5 + 4];
  if (lane == 0) { pz[0] = z0; pz[1] = z1; pz[2] = z2; pz[3] = z3; pz[4] = z4; }
  float prevT = 0.f;
  for (int base = 0; base < N_CELLS; base += 2048) {
    for (int q = lane; q < 2048; q += 32) sT[q] = Ts[base + q];
    __syncthreads();
    int t0 = (base == 0) ? 1 : 0;
    if (base == 0) prevT = sT[0];
    for (int t = t0; t < 2048; ++t) {
      float curT = sT[t];
      float dt = curT - prevT; prevT = curT;
      float aj = z0 * w10 + z1 * w11 + z2 * w12 + z3 * w13 + z4 * w14 + b1j;
      float e = (lane < N_ODEH) ? eluf(aj) : 0.f;
      float r0 = e * w20, r1 = e * w21, r2 = e * w22, r3 = e * w23, r4 = e * w24;
#pragma unroll
      for (int off = 16; off; off >>= 1) {
        r0 += __shfl_xor(r0, off, 32);
        r1 += __shfl_xor(r1, off, 32);
        r2 += __shfl_xor(r2, off, 32);
        r3 += __shfl_xor(r3, off, 32);
        r4 += __shfl_xor(r4, off, 32);
      }
      z0 += dt * (r0 + b20); z1 += dt * (r1 + b21); z2 += dt * (r2 + b22);
      z3 += dt * (r3 + b23); z4 += dt * (r4 + b24);
      if (lane == 0) {
        size_t o = (size_t)(base + t) * 5;
        pz[o + 0] = z0; pz[o + 1] = z1; pz[o + 2] = z2; pz[o + 3] = z3; pz[o + 4] = z4;
      }
    }
    __syncthreads();
  }
}

// ------- decoder hidden layers, written directly in A-fragment order + z_div -
// A-frag index: [tile][mt][ks][lane][e]; inverse of 16-bit A 16x32 layout
__global__ __launch_bounds__(128) void k_hd(const int* __restrict__ idx,
                                            const float* __restrict__ zarr,
                                            const float* __restrict__ pz,
                                            const float* __restrict__ w1,
                                            const float* __restrict__ b1,
                                            _Float16* __restrict__ hds,
                                            float* __restrict__ acc)
{
  const int s = blockIdx.x;
  const int j = threadIdx.x;
  const int ii = idx[s];
  float zs0 = zarr[(size_t)ii * 5 + 0], zs1 = zarr[(size_t)ii * 5 + 1],
        zs2 = zarr[(size_t)ii * 5 + 2], zs3 = zarr[(size_t)ii * 5 + 3],
        zs4 = zarr[(size_t)ii * 5 + 4];
  float p0 = pz[(size_t)s * 5 + 0], p1 = pz[(size_t)s * 5 + 1],
        p2 = pz[(size_t)s * 5 + 2], p3 = pz[(size_t)s * 5 + 3],
        p4 = pz[(size_t)s * 5 + 4];
  float bj = b1[j];
  float a1 = zs0 * w1[0 * N_VH + j] + zs1 * w1[1 * N_VH + j] + zs2 * w1[2 * N_VH + j]
           + zs3 * w1[3 * N_VH + j] + zs4 * w1[4 * N_VH + j] + bj;
  float a2 = p0 * w1[0 * N_VH + j] + p1 * w1[1 * N_VH + j] + p2 * w1[2 * N_VH + j]
           + p3 * w1[3 * N_VH + j] + p4 * w1[4 * N_VH + j] + bj;

  // invert A-fragment mapping: j = ks*32 + ((v>>2)<<4) + (g<<3) + ((v&3)<<1) + p
  const int tile = s >> 4, m = s & 15;
  const int ks = j >> 5, kp = j & 31;
  const int half = kp >> 4;
  const int rem  = kp & 15;
  const int g    = rem >> 3;
  const int r2   = rem & 7;
  const int v    = half * 4 + (r2 >> 1);
  const int e    = 2 * v + (r2 & 1);
  const int lane = g * 16 + m;
  const size_t b0 = ((((size_t)tile * 2 + 0) * 4 + ks) * 32 + lane) * 16 + e;
  const size_t b1i = ((((size_t)tile * 2 + 1) * 4 + ks) * 32 + lane) * 16 + e;
  hds[b0]  = (_Float16)eluf(a1);
  hds[b1i] = (_Float16)eluf(a2);

  if (j == 0) {
    float d0 = zs0 - p0, d1 = zs1 - p1, d2 = zs2 - p2, d3 = zs3 - p3, d4 = zs4 - p4;
    atomicAdd(acc + 1, d0 * d0 + d1 * d1 + d2 * d2 + d3 * d3 + d4 * d4);
  }
}

// ---------------- decoder GEMM + softmax + NB loss (3-pass WMMA) -------------
__global__ __launch_bounds__(256) void k_decoder(const float* __restrict__ x,
                                                 const float* __restrict__ y,
                                                 const _Float16* __restrict__ w2s,
                                                 const float* __restrict__ b2,
                                                 const int* __restrict__ idx,
                                                 const _Float16* __restrict__ hds,
                                                 const float* __restrict__ thetaA,
                                                 const float* __restrict__ ltA,
                                                 const float* __restrict__ lgA,
                                                 float* __restrict__ acc)
{
  __shared__ float sWaveRed[8][32];
  __shared__ float sRowMax[32];
  __shared__ float sRowSum[32];
  __shared__ int   sIdx[16];
  __shared__ float sY[16];

  const int tid  = threadIdx.x;
  const int lane = tid & 31;
  const int wave = tid >> 5;
  const int g    = lane >> 4;
  const int m    = lane & 15;
  const int s0   = blockIdx.x * 16;

  if (tid < 16) {
    int ii = idx[s0 + tid];
    sIdx[tid] = ii;
    sY[tid]   = y[ii];
  }
  __syncthreads();

  // register-resident A fragments, pre-swizzled by k_hd: 2 b128 loads each
  v16h af[2][4];
#pragma unroll
  for (int mt = 0; mt < 2; ++mt)
#pragma unroll
    for (int ks = 0; ks < 4; ++ks)
      af[mt][ks] = *(const v16h*)(hds +
          ((((size_t)blockIdx.x * 2 + mt) * 4 + ks) * 32 + lane) * 16);

  float lossEC = 0.f, lossODE = 0.f;
  const v8f zero = {0.f, 0.f, 0.f, 0.f, 0.f, 0.f, 0.f, 0.f};

  for (int pass = 0; pass < 3; ++pass) {
    float r0a[8], r1a[8];
#pragma unroll
    for (int v = 0; v < 8; ++v) {
      r0a[v] = (pass == 0) ? -3.4e38f : 0.f;
      r1a[v] = r0a[v];
    }

    for (int it = 0; it < 32; ++it) {
      const int nt  = it * 8 + wave;          // col tile (of 256)
      const int n   = nt * 16 + m;
      const float b2n = b2[n];
      v8f c0 = zero, c1 = zero;
#pragma unroll
      for (int ks = 0; ks < 4; ++ks) {
        v16h bf = *(const v16h*)(w2s + (((size_t)nt * 4 + ks) * 32 + lane) * 16);
        c0 = __builtin_amdgcn_wmma_f32_16x16x32_f16(false, af[0][ks], false, bf,
                                                    (short)0, c0, false, false);
        c1 = __builtin_amdgcn_wmma_f32_16x16x32_f16(false, af[1][ks], false, bf,
                                                    (short)0, c1, false, false);
      }
      if (pass == 0) {
#pragma unroll
        for (int v = 0; v < 8; ++v) {
          r0a[v] = fmaxf(r0a[v], c0[v] + b2n);
          r1a[v] = fmaxf(r1a[v], c1[v] + b2n);
        }
      } else if (pass == 1) {
#pragma unroll
        for (int v = 0; v < 8; ++v) {
          int R = v + (g << 3);
          r0a[v] += __expf(c0[v] + b2n - sRowMax[R]);
          r1a[v] += __expf(c1[v] + b2n - sRowMax[16 + R]);
        }
      } else {
        float th = thetaA[n], ltv = ltA[n], lgv = lgA[n];
#pragma unroll
        for (int v = 0; v < 8; ++v) {
          int cell = v + (g << 3);
          float mu1 = sY[cell] * __expf(c0[v] + b2n - sRowMax[cell]) / sRowSum[cell];
          float mu2 = sY[cell] * __expf(c1[v] + b2n - sRowMax[16 + cell]) / sRowSum[16 + cell];
          float xv = x[(size_t)sIdx[cell] * N_INT + n];
          float common = lgammaf(xv + th) - lgv - lgammaf(xv + 1.f) + th * ltv;
          lossEC  += common + xv * logf(mu1 + EPSF) - (th + xv) * logf(th + mu1 + EPSF);
          lossODE += common + xv * logf(mu2 + EPSF) - (th + xv) * logf(th + mu2 + EPSF);
        }
      }
    }

    if (pass < 2) {
#pragma unroll
      for (int v = 0; v < 8; ++v) {
        float a = r0a[v], b = r1a[v];
#pragma unroll
        for (int off = 1; off < 16; off <<= 1) {   // reduce within 16-lane half
          float as = __shfl_xor(a, off, 32);
          float bs = __shfl_xor(b, off, 32);
          if (pass == 0) { a = fmaxf(a, as); b = fmaxf(b, bs); }
          else           { a += as;          b += bs; }
        }
        if (m == 0) {
          sWaveRed[wave][v + (g << 3)]      = a;
          sWaveRed[wave][16 + v + (g << 3)] = b;
        }
      }
      __syncthreads();
      if (tid < 32) {
        float sres = (pass == 0) ? -3.4e38f : 0.f;
        for (int w = 0; w < 8; ++w) {
          float vv = sWaveRed[w][tid];
          sres = (pass == 0) ? fmaxf(sres, vv) : (sres + vv);
        }
        if (pass == 0) sRowMax[tid] = sres; else sRowSum[tid] = sres;
      }
      __syncthreads();
    }
  }

#pragma unroll
  for (int off = 16; off; off >>= 1) {
    lossEC  += __shfl_xor(lossEC, off, 32);
    lossODE += __shfl_xor(lossODE, off, 32);
  }
  if (lane == 0) {
    atomicAdd(acc + 2, lossEC);
    atomicAdd(acc + 3, lossODE);
  }
}

// ---------------- finalize ----------------------------------------------------
__global__ void k_final(const float* __restrict__ acc, float* __restrict__ out)
{
  const float inv = 1.f / (float)N_CELLS;
  float kl  = acc[0] * inv;
  float zd  = acc[1] * inv;
  float ec  = -acc[2] * inv;
  float ode = -acc[3] * inv;
  out[0] = 0.5f * ec + 0.5f * ode + zd + kl;
  out[1] = ec;
  out[2] = ode;
  out[3] = kl;
  out[4] = zd;
}

extern "C" void kernel_launch(void* const* d_in, const int* in_sizes, int n_in,
                              void* d_out, int out_size, void* d_ws, size_t ws_size,
                              hipStream_t stream)
{
  (void)in_sizes; (void)n_in; (void)out_size; (void)ws_size;
  const float* x     = (const float*)d_in[0];
  const float* y     = (const float*)d_in[1];
  const float* eps   = (const float*)d_in[2];
  const float* encW1 = (const float*)d_in[3];
  const float* encb1 = (const float*)d_in[4];
  const float* encW2 = (const float*)d_in[5];
  const float* encb2 = (const float*)d_in[6];
  const float* encW3 = (const float*)d_in[7];
  const float* encb3 = (const float*)d_in[8];
  const float* odeW1 = (const float*)d_in[9];
  const float* odeb1 = (const float*)d_in[10];
  const float* odeW2 = (const float*)d_in[11];
  const float* odeb2 = (const float*)d_in[12];
  const float* decW1 = (const float*)d_in[13];
  const float* decb1 = (const float*)d_in[14];
  const float* decW2 = (const float*)d_in[15];
  const float* decb2 = (const float*)d_in[16];
  const float* disp  = (const float*)d_in[17];

  float* ws  = (float*)d_ws;
  float* out = (float*)d_out;
  _Float16* w1s = (_Float16*)(ws + OFF_W1S);
  _Float16* w2s = (_Float16*)(ws + OFF_W2S);
  _Float16* hds = (_Float16*)(ws + OFF_HDS);

  k_init<<<16, 256, 0, stream>>>(disp, ws);
  k_swz_w1<<<128, 256, 0, stream>>>(encW1, w1s);
  k_swz_w2<<<128, 256, 0, stream>>>(decW2, w2s);
  k_encoder<<<512, 128, 0, stream>>>(x, w1s, encb1, ws + OFF_H);
  k_stats<<<N_CELLS / 256, 256, 0, stream>>>(ws + OFF_H, encW2, encb2, encW3, encb3,
                                             eps, ws + OFF_T, ws + OFF_Z, ws + OFF_ACC);
  k_rank<<<N_CELLS, 256, 0, stream>>>(ws + OFF_T, (int*)(ws + OFF_IDX), ws + OFF_TS);
  k_ode<<<1, 32, 0, stream>>>(odeW1, odeb1, odeW2, odeb2, ws + OFF_TS,
                              (const int*)(ws + OFF_IDX), ws + OFF_Z, ws + OFF_PZ);
  k_hd<<<N_CELLS, 128, 0, stream>>>((const int*)(ws + OFF_IDX), ws + OFF_Z, ws + OFF_PZ,
                                    decW1, decb1, hds, ws + OFF_ACC);
  k_decoder<<<N_CELLS / 16, 256, 0, stream>>>(x, y, w2s, decb2,
                                              (const int*)(ws + OFF_IDX), hds,
                                              ws + OFF_THETA, ws + OFF_LT, ws + OFF_LG,
                                              ws + OFF_ACC);
  k_final<<<1, 1, 0, stream>>>(ws + OFF_ACC, out);
}
